// MultiHeadAttention_51118700757519
// MI455X (gfx1250) — matrix operand order repaired and checked by reading, not compiled, not run on hardware
//
#include <hip/hip_runtime.h>
#include <hip/hip_bf16.h>
#include <math.h>

#define Bn 4
#define Sn 2048
#define En 512
#define Hn 8
#define Dn 64

typedef __attribute__((ext_vector_type(16))) _Float16 v16h;
typedef __attribute__((ext_vector_type(8)))  _Float16 v8h;
typedef __attribute__((ext_vector_type(4)))  _Float16 v4h;
typedef __attribute__((ext_vector_type(8)))  float    v8f;
typedef __attribute__((ext_vector_type(4)))  float    v4f;

// ---- WMMA fragment loaders (per ISA 7.12.2 layouts, wave32) ----
// A-matrix 16x32 f16: lane holds row = lane%16; VGPR0-3 = K(koff..koff+7),
// VGPR4-7 = K(koff+16..koff+23), koff = (lane<16)?0:8. Two 16B contiguous loads.
static __device__ inline v16h load_a16(const _Float16* base, int lda, int lane) {
  int row  = lane & 15;
  int koff = (lane < 16) ? 0 : 8;
  const _Float16* p = base + (size_t)row * lda + koff;
  v8h lo = *(const v8h*)(p);
  v8h hi = *(const v8h*)(p + 16);
  v16h a;
#pragma unroll
  for (int i = 0; i < 8; ++i) { a[i] = lo[i]; a[i + 8] = hi[i]; }
  return a;
}

// B-matrix 32x16 f16: lane holds col = lane%16; b[0..7] = B[kg..kg+7][col],
// b[8..15] = B[kg+16..kg+23][col], kg = (lane<16)?0:8. Two 16B loads from B^T.
static __device__ inline v16h load_b16(const _Float16* bt, int ldb, int lane) {
  int col = lane & 15;
  int kg  = (lane < 16) ? 0 : 8;
  const _Float16* p = bt + (size_t)col * ldb + kg;
  v8h lo = *(const v8h*)(p);
  v8h hi = *(const v8h*)(p + 16);
  v16h b;
#pragma unroll
  for (int i = 0; i < 8; ++i) { b[i] = lo[i]; b[i + 8] = hi[i]; }
  return b;
}

static __device__ inline v8f wmma_f16(v16h a, v16h b, v8f c) {
  return __builtin_amdgcn_wmma_f32_16x16x32_f16(false, a, false, b, (short)0, c,
                                                false, false);
}

// ---- fp32 -> f16 conversion (vectorized x4) ----
__global__ void cvt_f32_f16(const float* __restrict__ src, _Float16* __restrict__ dst, int n4) {
  int i = blockIdx.x * blockDim.x + threadIdx.x;
  if (i < n4) {
    v4f v = *(const v4f*)(src + (size_t)i * 4);
    v4h h;
#pragma unroll
    for (int j = 0; j < 4; ++j) h[j] = (_Float16)v[j];
    *(v4h*)(dst + (size_t)i * 4) = h;
  }
}

// ---- QKV projection: out = X @ W^T + b ; scatter into attention layouts ----
// 32x32 tile per wave (2 A-frags x 2 B-frags -> 4 WMMA per k-step, 16 FLOP/B).
// transposed==0 -> [B,H,S,D] (Q,K); transposed==1 -> [B,H,D,S] (V^T)
__global__ void gemm_qkv(const _Float16* __restrict__ X, const _Float16* __restrict__ W,
                         const float* __restrict__ bias, _Float16* __restrict__ out,
                         int transposed) {
  int lane = threadIdx.x & 31;
  int wave = threadIdx.x >> 5;
  int mt = blockIdx.x;                 // 256 row tiles of 32 (M = B*S = 8192)
  int nt = blockIdx.y * 4 + wave;      // 16 col tiles of 32 (N = E = 512)
  const _Float16* A  = X + (size_t)mt * 32 * En;
  const _Float16* Bt = W + (size_t)nt * 32 * En;   // W is (N x K) row-major = B^T
  v8f c00 = {}, c01 = {}, c10 = {}, c11 = {};
#pragma unroll 2
  for (int k = 0; k < En; k += 32) {
    v16h a0 = load_a16(A + k, En, lane);
    v16h a1 = load_a16(A + (size_t)16 * En + k, En, lane);
    v16h b0 = load_b16(Bt + k, En, lane);
    v16h b1 = load_b16(Bt + (size_t)16 * En + k, En, lane);
    c00 = wmma_f16(a0, b0, c00);
    c01 = wmma_f16(a0, b1, c01);
    c10 = wmma_f16(a1, b0, c10);
    c11 = wmma_f16(a1, b1, c11);
  }
  int col = lane & 15;
  int half = lane >> 4;
#pragma unroll
  for (int i = 0; i < 2; ++i) {
#pragma unroll
    for (int j = 0; j < 2; ++j) {
      v8f c = (i == 0) ? (j == 0 ? c00 : c01) : (j == 0 ? c10 : c11);
      int n = nt * 32 + j * 16 + col;
      float bn = bias[n];
      int hh = n / Dn, dd = n % Dn;
#pragma unroll
      for (int r = 0; r < 8; ++r) {
        int m  = mt * 32 + i * 16 + r + half * 8;
        int bb = m / Sn, ss = m % Sn;
        size_t idx = transposed
            ? ((size_t)(bb * Hn + hh) * Dn + dd) * Sn + ss
            : ((size_t)(bb * Hn + hh) * Sn + ss) * Dn + dd;
        out[idx] = (_Float16)(c[r] + bn);
      }
    }
  }
}

// ---- Fused attention: scores -> softmax (LDS) -> attn out + ctx = attn @ V ----
// block = 128 threads (4 waves); one block per (b, h, 16-row q tile)
__global__ void attn_kernel(const _Float16* __restrict__ Qh, const _Float16* __restrict__ Kh,
                            const _Float16* __restrict__ Vth, const float* __restrict__ pbias,
                            const int* __restrict__ msk, float* __restrict__ attn,
                            _Float16* __restrict__ ctxh) {
  extern __shared__ float smem[];
  float* sc   = smem;                  // 16 x 2048 scores / normalized attn
  float* redm = sc + 16 * Sn;          // 128
  float* reds = redm + 128;            // 128
  float* rowM = reds + 128;            // 16
  float* rowI = rowM + 16;             // 16

  int lane = threadIdx.x & 31;
  int wave = threadIdx.x >> 5;
  int qt = blockIdx.x, hh = blockIdx.y, bb = blockIdx.z;
  int col  = lane & 15;
  int half = lane >> 4;
  int ktend = qt + 1;                  // causal: only tiles with k_start <= q_end
  int kend  = ktend * 16;

  // Phase 1: score tiles (QK^T / sqrt(D) + bias, causal+pad mask) into LDS
  const _Float16* Qb = Qh + ((size_t)(bb * Hn + hh) * Sn + qt * 16) * Dn;
  v16h aq0 = load_a16(Qb, Dn, lane);        // d = 0..31
  v16h aq1 = load_a16(Qb + 32, Dn, lane);   // d = 32..63
  for (int kt = wave; kt < ktend; kt += 4) {
    const _Float16* Kb = Kh + ((size_t)(bb * Hn + hh) * Sn + kt * 16) * Dn;
    v16h b0 = load_b16(Kb, Dn, lane);
    v16h b1 = load_b16(Kb + 32, Dn, lane);
    v8f c = {};
    c = wmma_f16(aq0, b0, c);
    c = wmma_f16(aq1, b1, c);
    int kidx = kt * 16 + col;
    int mv = msk[bb * Sn + kidx];
    const float* bp = pbias + ((size_t)hh * Sn + qt * 16) * Sn + kidx;
#pragma unroll
    for (int r = 0; r < 8; ++r) {
      int rr = r + half * 8;
      int q  = qt * 16 + rr;
      float v = c[r] * 0.125f + bp[(size_t)rr * Sn];   // 1/sqrt(64)
      if (kidx > q || mv == 0) v = -__builtin_inff();
      sc[rr * Sn + kidx] = v;
    }
  }
  __syncthreads();

  // Phase 2: row-wise softmax stats (8 threads per row), then normalize,
  // writing fp32 attn to HBM (non-temporal: 537 MB streamed once, keep L2
  // for K/V/bias) and normalized values (incl. zeros) back to LDS.
  int tid = threadIdx.x;
  int rr = tid & 15;
  int ch = tid >> 4;
  float mloc = -__builtin_inff();
  for (int k = ch; k < kend; k += 8) mloc = fmaxf(mloc, sc[rr * Sn + k]);
  redm[rr * 8 + ch] = mloc;
  __syncthreads();
  float rowm = redm[rr * 8];
#pragma unroll
  for (int j = 1; j < 8; ++j) rowm = fmaxf(rowm, redm[rr * 8 + j]);
  float sloc = 0.f;
  for (int k = ch; k < kend; k += 8) sloc += __expf(sc[rr * Sn + k] - rowm);
  reds[rr * 8 + ch] = sloc;
  __syncthreads();
  float rsum = 0.f;
#pragma unroll
  for (int j = 0; j < 8; ++j) rsum += reds[rr * 8 + j];
  if (ch == 0) { rowM[rr] = rowm; rowI[rr] = 1.0f / rsum; }
  __syncthreads();

  size_t abase = ((size_t)(bb * Hn + hh) * Sn + qt * 16) * Sn;
  for (int idx = tid; idx < 16 * Sn; idx += 128) {
    int r2 = idx >> 11;            // / 2048
    int c2 = idx & (Sn - 1);
    int q  = qt * 16 + r2;
    float v = 0.f;
    if (c2 <= q) v = __expf(sc[r2 * Sn + c2] - rowM[r2]) * rowI[r2];
    sc[r2 * Sn + c2] = v;
    __builtin_nontemporal_store(v, &attn[abase + (size_t)r2 * Sn + c2]);
  }
  __syncthreads();

  // Phase 3: ctx = attn @ V ; each wave owns one 16-wide d stripe
  int nk = (kend + 31) >> 5;
  v8f cc = {};
  const _Float16* Vb = Vth + ((size_t)(bb * Hn + hh) * Dn + wave * 16) * Sn; // [B,H,D,S]
  int arow = lane & 15;
  int koff = (lane < 16) ? 0 : 8;
  for (int j = 0; j < nk; ++j) {
    const float* ap = sc + arow * Sn + j * 32 + koff;
    v4f x0 = *(const v4f*)(ap);
    v4f x1 = *(const v4f*)(ap + 4);
    v4f x2 = *(const v4f*)(ap + 16);
    v4f x3 = *(const v4f*)(ap + 20);
    v16h a;
#pragma unroll
    for (int i = 0; i < 4; ++i) {
      a[i]      = (_Float16)x0[i];
      a[4 + i]  = (_Float16)x1[i];
      a[8 + i]  = (_Float16)x2[i];
      a[12 + i] = (_Float16)x3[i];
    }
    v16h bf = load_b16(Vb + j * 32, Sn, lane);
    cc = wmma_f16(a, bf, cc);
  }
#pragma unroll
  for (int r = 0; r < 8; ++r) {
    int q = qt * 16 + r + half * 8;
    size_t o = ((size_t)bb * Sn + q) * En + hh * Dn + wave * 16 + col; // [B,S,(h,d)]
    ctxh[o] = (_Float16)cc[r];
  }
}

// ---- Output projection: out = ctx @ Wo^T + bo (fp32 result), 32x32/wave ----
__global__ void gemm_out(const _Float16* __restrict__ Ctx, const _Float16* __restrict__ Wo,
                         const float* __restrict__ bo, float* __restrict__ out) {
  int lane = threadIdx.x & 31;
  int wave = threadIdx.x >> 5;
  int mt = blockIdx.x;
  int nt = blockIdx.y * 4 + wave;
  const _Float16* A  = Ctx + (size_t)mt * 32 * En;
  const _Float16* Bt = Wo + (size_t)nt * 32 * En;
  v8f c00 = {}, c01 = {}, c10 = {}, c11 = {};
#pragma unroll 2
  for (int k = 0; k < En; k += 32) {
    v16h a0 = load_a16(A + k, En, lane);
    v16h a1 = load_a16(A + (size_t)16 * En + k, En, lane);
    v16h b0 = load_b16(Bt + k, En, lane);
    v16h b1 = load_b16(Bt + (size_t)16 * En + k, En, lane);
    c00 = wmma_f16(a0, b0, c00);
    c01 = wmma_f16(a0, b1, c01);
    c10 = wmma_f16(a1, b0, c10);
    c11 = wmma_f16(a1, b1, c11);
  }
  int col = lane & 15;
  int half = lane >> 4;
#pragma unroll
  for (int i = 0; i < 2; ++i) {
#pragma unroll
    for (int j = 0; j < 2; ++j) {
      v8f c = (i == 0) ? (j == 0 ? c00 : c01) : (j == 0 ? c10 : c11);
      int n = nt * 32 + j * 16 + col;
      float bn = bo[n];
#pragma unroll
      for (int r = 0; r < 8; ++r) {
        int m = mt * 32 + i * 16 + r + half * 8;
        out[(size_t)m * En + n] = c[r] + bn;
      }
    }
  }
}

extern "C" void kernel_launch(void* const* d_in, const int* in_sizes, int n_in,
                              void* d_out, int out_size, void* d_ws, size_t ws_size,
                              hipStream_t stream) {
  (void)in_sizes; (void)n_in; (void)out_size; (void)ws_size;
  const float* query = (const float*)d_in[0];
  const float* key_  = (const float*)d_in[1];
  const float* value = (const float*)d_in[2];
  const int*   mask  = (const int*)d_in[3];
  const float* pbias = (const float*)d_in[4];
  const float* Wq = (const float*)d_in[5];
  const float* bq = (const float*)d_in[6];
  const float* Wk = (const float*)d_in[7];
  const float* bk = (const float*)d_in[8];
  const float* Wv = (const float*)d_in[9];
  const float* bv = (const float*)d_in[10];
  const float* Wo = (const float*)d_in[11];
  const float* bo = (const float*)d_in[12];

  float* out_f  = (float*)d_out;
  float* attn_f = out_f + (size_t)Bn * Sn * En;   // outputs concatenated: (out, attn)

  // Workspace carve-up (f16): 3 converted inputs, 4 converted weights,
  // Q [B,H,S,D], K [B,H,S,D], V^T [B,H,D,S], ctx [B,S,E]  -> ~58 MB total
  const size_t NX = (size_t)Bn * Sn * En;   // 4,194,304
  const size_t NW = (size_t)En * En;        // 262,144
  _Float16* Xq  = (_Float16*)d_ws;
  _Float16* Xk  = Xq  + NX;
  _Float16* Xv  = Xk  + NX;
  _Float16* Wqh = Xv  + NX;
  _Float16* Wkh = Wqh + NW;
  _Float16* Wvh = Wkh + NW;
  _Float16* Woh = Wvh + NW;
  _Float16* Qh  = Woh + NW;
  _Float16* Kh  = Qh  + NX;
  _Float16* Vth = Kh  + NX;
  _Float16* Ctx = Vth + NX;

  int n4x = (int)(NX / 4);
  int n4w = (int)(NW / 4);
  cvt_f32_f16<<<dim3(n4x / 256), dim3(256), 0, stream>>>(query, Xq, n4x);
  cvt_f32_f16<<<dim3(n4x / 256), dim3(256), 0, stream>>>(key_,  Xk, n4x);
  cvt_f32_f16<<<dim3(n4x / 256), dim3(256), 0, stream>>>(value, Xv, n4x);
  cvt_f32_f16<<<dim3(n4w / 256), dim3(256), 0, stream>>>(Wq, Wqh, n4w);
  cvt_f32_f16<<<dim3(n4w / 256), dim3(256), 0, stream>>>(Wk, Wkh, n4w);
  cvt_f32_f16<<<dim3(n4w / 256), dim3(256), 0, stream>>>(Wv, Wvh, n4w);
  cvt_f32_f16<<<dim3(n4w / 256), dim3(256), 0, stream>>>(Wo, Woh, n4w);

  // QKV projections: M tiles = 8192/32 = 256, N tiles = 512/32 = 16 (4/block)
  gemm_qkv<<<dim3(256, 4), dim3(128), 0, stream>>>(Xq, Wqh, bq, Qh, 0);
  gemm_qkv<<<dim3(256, 4), dim3(128), 0, stream>>>(Xk, Wkh, bk, Kh, 0);
  gemm_qkv<<<dim3(256, 4), dim3(128), 0, stream>>>(Xv, Wvh, bv, Vth, 1);

  // Fused attention: one block per (q-tile, head, batch); 132 KB dynamic LDS
  // (CDNA5 WGP has 320 KB LDS; raise the per-kernel cap accordingly)
  size_t smem = (size_t)(16 * Sn + 288) * sizeof(float);
  (void)hipFuncSetAttribute((const void*)attn_kernel,
                            hipFuncAttributeMaxDynamicSharedMemorySize, (int)smem);
  attn_kernel<<<dim3(Sn / 16, Hn, Bn), dim3(128), smem, stream>>>(
      Qh, Kh, Vth, pbias, mask, attn_f, Ctx);

  // Output projection
  gemm_out<<<dim3(256, 4), dim3(128), 0, stream>>>(Ctx, Woh, bo, out_f);
}
